// MemoryUnit_57990648430879
// MI455X (gfx1250) — matrix operand order repaired and checked by reading, not compiled, not run on hardware
//
#include <hip/hip_runtime.h>
#include <hip/hip_bf16.h>

#define B_ROWS   32768
#define FEA      2048
#define NBANK    20
#define SHRINKL  0.0025f
#define LOG2E    1.4426950408889634f

typedef __attribute__((ext_vector_type(16))) _Float16 v16h;
typedef __attribute__((ext_vector_type(8)))  float    v8f;
typedef __attribute__((ext_vector_type(2)))  _Float16 h2;

struct U4 { unsigned x, y, z, w; };

union Frag {
    v16h v;
    h2   h[8];
    U4   q[2];
};

union PkU { h2 h; unsigned u; };

__device__ __forceinline__ h2 pk_f16(float a, float b) {
#if __has_builtin(__builtin_amdgcn_cvt_pkrtz)
    auto t = __builtin_amdgcn_cvt_pkrtz(a, b);   // v_cvt_pk_rtz_f16_f32
    return __builtin_bit_cast(h2, t);
#else
    h2 r; r[0] = (_Float16)a; r[1] = (_Float16)b; return r;
#endif
}

__device__ __forceinline__ float fast_exp2(float x) {
#if __has_builtin(__builtin_amdgcn_exp2f)
    return __builtin_amdgcn_exp2f(x);            // v_exp_f32
#else
    return exp2f(x);
#endif
}

__device__ __forceinline__ float fast_tanh(float x) {
#if __has_builtin(__builtin_amdgcn_tanhf)
    return __builtin_amdgcn_tanhf(x);            // v_tanh_f32 (gfx1250 transcendental)
#elif __has_builtin(__builtin_amdgcn_tanh_f32)
    return __builtin_amdgcn_tanh_f32(x);
#else
    float e = fast_exp2(x * (2.0f * LOG2E));     // tanh = 1 - 2/(e^(2x)+1)
    return 1.0f - 2.0f / (e + 1.0f);
#endif
}

__device__ __forceinline__ void wait_lds() {
#if __has_builtin(__builtin_amdgcn_s_wait_dscnt)
    __builtin_amdgcn_s_wait_dscnt(0);
#else
    asm volatile("s_wait_dscnt 0" ::: "memory");
#endif
}

// LDS layout (dwords):
//   [0, 32768)      : GEMM2 B-fragments: (nc*32 + lane)*8 + v  -> f16x2 of bank[k..k+1][nc*16 + (lane&15)]
//   [32768, 36864)  : per-wave att scratch: wave*512 + row*32 + n (float)
#define LDS_B2_DW   32768
#define LDS_ATT_DW  4096

__global__ void MemoryUnit_57990648430879_kernel(const float* __restrict__ x,
                                                 const float* __restrict__ bank,
                                                 float* __restrict__ out) {
    __shared__ unsigned s_lds[LDS_B2_DW + LDS_ATT_DW];   // 144 KB

    const int tid  = threadIdx.x;
    const int lane = tid & 31;
    const int w    = tid >> 5;      // wave id 0..7
    const int hi   = lane >> 4;     // half-wave select
    const int ln   = lane & 15;

    // ---- Pre-swizzle bank into GEMM2 WMMA B-fragments (f16), K=20 zero-padded to 32 ----
    // B order used here (contiguous k, same as A): VGPR v holds K = hi*16 + 2v, 2v+1 ; N = lane&15.
    // (K-padding zeros here line up with the zeroed att pad entries n = 20..31 in GEMM2.)
    for (int f = tid; f < LDS_B2_DW; f += 256) {
        const int v  = f & 7;
        const int l  = (f >> 3) & 31;
        const int nc = f >> 8;
        const int n  = nc * 16 + (l & 15);
        const int k  = ((l >> 4) << 4) + (v << 1);
        const float b0 = (k     < NBANK) ? bank[(size_t)k       * FEA + n] : 0.0f;
        const float b1 = (k + 1 < NBANK) ? bank[(size_t)(k + 1) * FEA + n] : 0.0f;
        PkU p; p.h = pk_f16(b0, b1);
        s_lds[f] = p.u;
    }
    __syncthreads();

    const int rowTile = blockIdx.x * 128 + w * 16;
    const float* xrow = x + (size_t)(rowTile + ln) * FEA;        // A row = lane&15
    const float* bnk0 = bank + (size_t)ln * FEA;                 // B tile0 row n = ln (<20)
    // Pad lanes (n = 20..31) read a CLAMPED valid row instead of being exec-masked:
    // a garbage B column only corrupts its own (discarded) C column, so no branches,
    // no zero-fills, no exec saves in the hot loop.
    const float* bnk1 = bank + (size_t)(16 + (ln & 3)) * FEA;    // rows 16..19, always in-bounds
    const bool   nv1  = (ln < (NBANK - 16));                     // n = 16..19 valid in C1

    // ---- GEMM1: logits[16,32] = x_tile[16,2048] @ bank^T, via 2 WMMA accumulators ----
    v8f acc0 = {};
    v8f acc1 = {};
    for (int kc = 0; kc < 64; ++kc) {
        // A fragment in this code's contiguous k order (matches B): K = hi*16 + 2v{,+1}.
        const int ka = kc * 32 + hi * 16;
        if (kc + 1 < 64) __builtin_prefetch(xrow + ka + 32, 0, 3);  // pull next chunk close
        const float4 a0 = *(const float4*)(xrow + ka);
        const float4 a1 = *(const float4*)(xrow + ka + 4);
        const float4 a2 = *(const float4*)(xrow + ka + 8);
        const float4 a3 = *(const float4*)(xrow + ka + 12);
        Frag A;
        A.h[0] = pk_f16(a0.x, a0.y); A.h[1] = pk_f16(a0.z, a0.w);
        A.h[2] = pk_f16(a1.x, a1.y); A.h[3] = pk_f16(a1.z, a1.w);
        A.h[4] = pk_f16(a2.x, a2.y); A.h[5] = pk_f16(a2.z, a2.w);
        A.h[6] = pk_f16(a3.x, a3.y); A.h[7] = pk_f16(a3.z, a3.w);

        // B fragments straight from bank (resident in L2/WGP$): K = kc*32 + hi*16 + 2v{,+1}
        const int kb = kc * 32 + hi * 16;
        Frag B0, B1;
        #pragma unroll
        for (int v = 0; v < 4; ++v) {
            const float4 t = *(const float4*)(bnk0 + kb + 4 * v);
            B0.h[2 * v]     = pk_f16(t.x, t.y);
            B0.h[2 * v + 1] = pk_f16(t.z, t.w);
            const float4 u = *(const float4*)(bnk1 + kb + 4 * v);  // unconditional, branchless
            B1.h[2 * v]     = pk_f16(u.x, u.y);
            B1.h[2 * v + 1] = pk_f16(u.z, u.w);
        }
        acc0 = __builtin_amdgcn_wmma_f32_16x16x32_f16(false, A.v, false, B0.v, (short)0, acc0, false, false);
        acc1 = __builtin_amdgcn_wmma_f32_16x16x32_f16(false, A.v, false, B1.v, (short)0, acc1, false, false);
    }

    // ---- softmax -> softshrink -> softmax over N=20, per row, across half-wave lanes ----
    // C layout: VGPR r, lane: M = r + hi*8, N = lane&15 (acc0: N 0..15, acc1: N 16..31).
    float* att = (float*)(s_lds + LDS_B2_DW) + w * 512;           // [16 rows][32 n]
    const float NEGINF = -__builtin_inff();
    #pragma unroll
    for (int r = 0; r < 8; ++r) {
        const float c0 = acc0[r];
        const float c1 = acc1[r];
        // softmax #1
        float m = fmaxf(c0, nv1 ? c1 : NEGINF);
        m = fmaxf(m, __shfl_xor(m, 1, 32));
        m = fmaxf(m, __shfl_xor(m, 2, 32));
        m = fmaxf(m, __shfl_xor(m, 4, 32));
        m = fmaxf(m, __shfl_xor(m, 8, 32));
        const float e0 = fast_exp2((c0 - m) * LOG2E);
        const float e1 = nv1 ? fast_exp2((c1 - m) * LOG2E) : 0.0f;
        float s = e0 + e1;
        s += __shfl_xor(s, 1, 32);
        s += __shfl_xor(s, 2, 32);
        s += __shfl_xor(s, 4, 32);
        s += __shfl_xor(s, 8, 32);
        const float inv = 1.0f / s;
        const float p0 = e0 * inv, p1 = e1 * inv;
        // softshrink (probabilities are >= 0, so only the upper branch can fire)
        const float t0 = (p0 > SHRINKL) ? (p0 - SHRINKL) : 0.0f;
        const float t1 = (p1 > SHRINKL) ? (p1 - SHRINKL) : 0.0f;
        // softmax #2 (zeros are valid entries among the 20)
        float m2 = fmaxf(t0, nv1 ? t1 : NEGINF);
        m2 = fmaxf(m2, __shfl_xor(m2, 1, 32));
        m2 = fmaxf(m2, __shfl_xor(m2, 2, 32));
        m2 = fmaxf(m2, __shfl_xor(m2, 4, 32));
        m2 = fmaxf(m2, __shfl_xor(m2, 8, 32));
        const float f0 = fast_exp2((t0 - m2) * LOG2E);
        const float f1 = nv1 ? fast_exp2((t1 - m2) * LOG2E) : 0.0f;
        float s2 = f0 + f1;
        s2 += __shfl_xor(s2, 1, 32);
        s2 += __shfl_xor(s2, 2, 32);
        s2 += __shfl_xor(s2, 4, 32);
        s2 += __shfl_xor(s2, 8, 32);
        const float inv2 = 1.0f / s2;
        const int M = r + hi * 8;
        att[M * 32 + ln]      = f0 * inv2;                 // n = 0..15
        att[M * 32 + 16 + ln] = nv1 ? (f1 * inv2) : 0.0f;  // n = 16..31 (pad -> 0 = GEMM2 K-pad)
    }
    wait_lds();   // wave-local LDS RAW (transpose through att scratch)

    // ---- Build GEMM2 A-fragment (16-bit A 16x32): M = lane&15, K = att index (padded to 32) ----
    Frag A2;
    #pragma unroll
    for (int v = 0; v < 4; ++v) {
        const float2 t0 = *(const float2*)&att[ln * 32 + hi * 16 + 2 * v];
        A2.h[v]     = pk_f16(t0.x, t0.y);
        const float2 t1 = *(const float2*)&att[ln * 32 + hi * 16 + 8 + 2 * v];
        A2.h[4 + v] = pk_f16(t1.x, t1.y);
    }

    // ---- GEMM2: out[16,2048] = att[16,32pad] @ bank, 128 N-chunks, tanh epilogue ----
    float* outRow = out + (size_t)rowTile * FEA;
    for (int nc = 0; nc < 128; ++nc) {
        Frag B2;
        const U4* p = (const U4*)&s_lds[(size_t)(nc * 32 + lane) * 8];
        B2.q[0] = p[0];                                     // ds_load_b128 x2
        B2.q[1] = p[1];
        v8f d = {};
        d = __builtin_amdgcn_wmma_f32_16x16x32_f16(false, A2.v, false, B2.v, (short)0, d, false, false);
        const int col = nc * 16 + ln;
        #pragma unroll
        for (int r = 0; r < 8; ++r) {
            outRow[(size_t)(r + hi * 8) * FEA + col] = fast_tanh(d[r]);
        }
    }
}

extern "C" void kernel_launch(void* const* d_in, const int* in_sizes, int n_in,
                              void* d_out, int out_size, void* d_ws, size_t ws_size,
                              hipStream_t stream) {
    (void)in_sizes; (void)n_in; (void)out_size; (void)d_ws; (void)ws_size;
    const float* x    = (const float*)d_in[0];   // [32768, 2048] f32
    const float* bank = (const float*)d_in[1];   // [20, 2048]    f32
    float* out        = (float*)d_out;           // [32768, 2048] f32

    dim3 grid(B_ROWS / 128);   // 256 blocks, 128 rows each
    dim3 block(256);           // 8 wave32 -> 8 independent 16-row WMMA tiles
    MemoryUnit_57990648430879_kernel<<<grid, block, 0, stream>>>(x, bank, out);
}
